// SoftVectorQuantizer_54674933678222
// MI455X (gfx1250) — matrix operand order repaired and checked by reading, not compiled, not run on hardware
//
#include <hip/hip_runtime.h>
#include <stdint.h>

#define N_ROWS 131072
#define DIM 256
#define KENT 1024
#define CHUNK 32
#define ROWS_PER_WG 128
#define NCHUNK (KENT / CHUNK)

typedef __attribute__((ext_vector_type(16))) __bf16 v16bf;
typedef __attribute__((ext_vector_type(8)))  float  v8f;
typedef uint32_t u32;
typedef uint16_t u16;
typedef __attribute__((ext_vector_type(4))) u32 u32x4;
typedef __attribute__((ext_vector_type(8))) int i32x8;
typedef __attribute__((ext_vector_type(4))) int i32x4;

#if __has_builtin(__builtin_amdgcn_tensor_load_to_lds)
#define HAVE_TDM 1
#if __has_include(<hip/amd_detail/amd_gfx1250_TDM.h>)
#define TDM_6ARG 1
#endif
#endif

union BF16x16 { v16bf v; uint4 q[2]; u32 w[8]; };

__device__ __forceinline__ u16 f32_to_bf16(float f) {
    u32 b = __float_as_uint(f);
    return (u16)((b + 0x7FFFu + ((b >> 16) & 1u)) >> 16);
}
__device__ __forceinline__ float bf16_to_f32(u16 h) {
    return __uint_as_float(((u32)h) << 16);
}

// Workspace layout (bytes): enorm f32[1024] | cbRM bf16[1024][256] | cbT bf16[256][1024]
#define WS_ENORM 0
#define WS_CBRM  4096
#define WS_CBT   (4096 + 524288)

// LDS layout (bytes)
#define OFF_ZBF   0          // 128*256 bf16 = 65536
#define OFF_RM0   65536      // 32*256 bf16  = 16384 (row-major [ent][dim])
#define OFF_RM1   81920
#define OFF_TR0   98304      // 256*32 bf16  = 16384 (transposed [dim][ent])
#define OFF_TR1   114688
#define OFF_ENS   131072     // 1024 f32     = 4096
#define LDS_BYTES 135168

#ifdef HAVE_TDM
// Issue a TDM 2D tile load per CDNA5 ISA.
// elem_code: 0=1B,1=2B,2=4B. Dims/strides in element units.
__device__ __forceinline__ void tdm_load_2d(const void* gsrc, u32 lds_off, u32 elem_code,
                                            u32 tensor_d0, u32 tensor_d1,
                                            u32 tile_d0, u32 tile_d1, u32 stride0) {
    unsigned long long ga = (unsigned long long)(uintptr_t)gsrc;
    u32x4 g0;
    g0[0] = 1u;                                   // count=1, is_restore=0, gather off
    g0[1] = lds_off;                              // lds_addr (bytes)
    g0[2] = (u32)ga;                              // global_addr[31:0]
    g0[3] = (u32)(ga >> 32) | (2u << 30);         // global_addr[56:32] | type=2 (image)
    i32x8 g1;
    g1[0] = (int)(elem_code << 16);               // wg_mask=0 | data_size
    g1[1] = (int)(tensor_d0 << 16);               // tensor_dim0[15:0] @ bits 63:48
    g1[2] = (int)(((tensor_d0 >> 16) & 0xFFFFu) | (tensor_d1 << 16));
    g1[3] = (int)(((tensor_d1 >> 16) & 0xFFFFu) | (tile_d0 << 16));
    g1[4] = (int)tile_d1;                         // tile_dim1 (tile_dim2=0)
    g1[5] = (int)stride0;                         // tensor_dim0_stride[31:0]
    g1[6] = 0;
    g1[7] = 0;
    i32x4 z4 = {0, 0, 0, 0};
#ifdef TDM_6ARG
    i32x8 z8 = {0, 0, 0, 0, 0, 0, 0, 0};
    __builtin_amdgcn_tensor_load_to_lds(g0, g1, z4, z4, z8, 0);
#else
    __builtin_amdgcn_tensor_load_to_lds(g0, g1, z4, z4, 0);
#endif
}
#endif

// prep (once per launch): enorm, bf16 codebook in both operand layouts, zero loss.
__global__ void svq_prep_kernel(const float* __restrict__ cb,
                                char* __restrict__ ws,
                                float* __restrict__ loss) {
    int k = blockIdx.x * blockDim.x + threadIdx.x;   // 0..1023 = codebook entry
    if (k == 0) *loss = 0.0f;
    if (k >= KENT) return;
    float* enorm = (float*)(ws + WS_ENORM);
    u16*   cbRM  = (u16*)(ws + WS_CBRM);
    u16*   cbT   = (u16*)(ws + WS_CBT);
    const float* row = cb + (size_t)k * DIM;
    float s = 0.f;
    u32* rmw = (u32*)(cbRM + (size_t)k * DIM);
    #pragma unroll 4
    for (int d = 0; d < DIM; d += 2) {
        float a = row[d], b = row[d + 1];
        s += a * a + b * b;
        u16 ha = f32_to_bf16(a), hb = f32_to_bf16(b);
        rmw[d >> 1] = (u32)ha | ((u32)hb << 16);
        cbT[(size_t)d * KENT + k]       = ha;
        cbT[(size_t)(d + 1) * KENT + k] = hb;
    }
    enorm[k] = s;
}

__global__ __launch_bounds__(256)
void svq_main_kernel(const float* __restrict__ z,
                     const char* __restrict__ ws,
                     float* __restrict__ out,
                     float* __restrict__ loss) {
    extern __shared__ char smem[];
    u16*   zbf = (u16*)(smem + OFF_ZBF);
    float* ens = (float*)(smem + OFF_ENS);
    const float* enorm = (const float*)(ws + WS_ENORM);
    const u16*   cbRM  = (const u16*)(ws + WS_CBRM);
    const u16*   cbT   = (const u16*)(ws + WS_CBT);

    const int lane = threadIdx.x & 31;
    const int wave = threadIdx.x >> 5;
    const int col  = lane & 15;     // z-row within wave tile (N-col of S^T and acc)
    const int hi   = lane >> 4;     // lane half

    // ---- stage this WG's 128 z rows to LDS as bf16 (coalesced); stage all enorm ----
    const float* zg = z + (size_t)blockIdx.x * ROWS_PER_WG * DIM;
    #pragma unroll 4
    for (int k = 0; k < 32; ++k) {
        int i4 = threadIdx.x + 256 * k;
        float4 f = ((const float4*)zg)[i4];
        uint2 p;
        p.x = (u32)f32_to_bf16(f.x) | ((u32)f32_to_bf16(f.y) << 16);
        p.y = (u32)f32_to_bf16(f.z) | ((u32)f32_to_bf16(f.w) << 16);
        ((uint2*)zbf)[i4] = p;
    }
    #pragma unroll
    for (int k = 0; k < 4; ++k) ens[threadIdx.x + 256 * k] = enorm[threadIdx.x + 256 * k];
    __syncthreads();

    // ---- per-wave B operand (z^T): lane holds z-row `col`, elems = dims 32s+8hi..+7, 32s+16+8hi..+7 ----
    const int rowL = wave * 16 + col;
    v16bf Bz[8];
    #pragma unroll
    for (int s = 0; s < 8; ++s) {
        const uint4* p = (const uint4*)(zbf + rowL * DIM + 32 * s + 8 * hi);
        BF16x16 t; t.q[0] = p[0]; t.q[1] = p[2];
        Bz[s] = t.v;
    }

    v8f acc[16];
    const v8f zero8 = {0.f,0.f,0.f,0.f,0.f,0.f,0.f,0.f};
    #pragma unroll
    for (int j = 0; j < 16; ++j) acc[j] = zero8;
    float m = -1e30f, lsum = 0.f;

#ifdef HAVE_TDM
    if (wave == 0) {   // prologue: chunk 0 tiles -> buffers 0
        tdm_load_2d(cbRM, OFF_RM0, 1u, DIM, KENT, DIM, CHUNK, DIM);
        tdm_load_2d(cbT,  OFF_TR0, 1u, KENT, DIM, CHUNK, DIM, KENT);
    }
#endif

    for (int kc = 0; kc < NCHUNK; ++kc) {
        const int base = kc * CHUNK;
        const u16* rmb = (const u16*)(smem + (((kc & 1) == 0) ? OFF_RM0 : OFF_RM1));
        const u16* trb = (const u16*)(smem + (((kc & 1) == 0) ? OFF_TR0 : OFF_TR1));

#ifdef HAVE_TDM
        if (wave == 0) __builtin_amdgcn_s_wait_tensorcnt((short)0);  // chunk kc tiles in LDS
        __syncthreads();   // publish; all waves also done reading the other buffers
        if (wave == 0 && kc + 1 < NCHUNK) {
            u32 rmo = (((kc + 1) & 1) == 0) ? OFF_RM0 : OFF_RM1;
            u32 tro = (((kc + 1) & 1) == 0) ? OFF_TR0 : OFF_TR1;
            tdm_load_2d(cbRM + (size_t)(base + CHUNK) * DIM, rmo, 1u, DIM, KENT, DIM, CHUNK, DIM);
            tdm_load_2d(cbT + (base + CHUNK), tro, 1u, KENT, DIM, CHUNK, DIM, KENT);
        }
#else
        __syncthreads();
        {   // synchronous fill of both tile buffers from pre-converted workspace
            uint4* drm = (uint4*)rmb;
            const uint4* srm = (const uint4*)(cbRM + (size_t)base * DIM);
            #pragma unroll
            for (int k = 0; k < 4; ++k) drm[threadIdx.x + 256 * k] = srm[threadIdx.x + 256 * k];
            // transposed tile: 256 rows x 32 entries, global stride KENT
            uint2* dtr = (uint2*)trb;
            int drow = threadIdx.x >> 2;          // wait: 256 threads, 256 rows, 4 uint2 per row
            int dseg = threadIdx.x & 3;
            const uint2* str = (const uint2*)(cbT + (size_t)0 + base);
            // each row: 32 u16 = 8 uint2; 256 rows * 8 = 2048 uint2 / 256 thr = 8 each
            #pragma unroll
            for (int k = 0; k < 8; ++k) {
                int idx = threadIdx.x + 256 * k;  // uint2 index in tile
                int r = idx >> 3, c = idx & 7;    // row, uint2-within-row
                dtr[idx] = *(const uint2*)(cbT + (size_t)r * KENT + base + c * 4);
            }
            (void)drow; (void)dseg; (void)str;
        }
        __syncthreads();
#endif

        // ---- GEMM1: S^T = E_chunk (16x32 A-tiles) x z^T (32x16 B) ----
        v8f s0 = zero8, s1 = zero8;
        #pragma unroll
        for (int s = 0; s < 8; ++s) {
            BF16x16 a0, a1;
            const u16* b0 = rmb + col * DIM        + 32 * s + 8 * hi;
            const u16* b1 = rmb + (col + 16) * DIM + 32 * s + 8 * hi;
            a0.q[0] = *(const uint4*)(b0);  a0.q[1] = *(const uint4*)(b0 + 16);
            a1.q[0] = *(const uint4*)(b1);  a1.q[1] = *(const uint4*)(b1 + 16);
            s0 = __builtin_amdgcn_wmma_f32_16x16x32_bf16(false, a0.v, false, Bz[s],
                                                         (short)0, s0, false, false);
            s1 = __builtin_amdgcn_wmma_f32_16x16x32_bf16(false, a1.v, false, Bz[s],
                                                         (short)0, s1, false, false);
        }

        // ---- online softmax over this chunk (lane owns z-row `col`) ----
        float l0[8], l1[8];
        #pragma unroll
        for (int r = 0; r < 8; ++r) {
            l0[r] = 2.f * s0[r] - ens[base + 8 * hi + r];
            l1[r] = 2.f * s1[r] - ens[base + 16 + 8 * hi + r];
        }
        float cm = l0[0];
        #pragma unroll
        for (int r = 0; r < 8; ++r) { cm = fmaxf(cm, l0[r]); cm = fmaxf(cm, l1[r]); }
        cm = fmaxf(cm, __shfl_xor(cm, 16, 32));
        float nm = fmaxf(m, cm);
        float scale = __expf(m - nm);
        float p0[8], p1[8], ps = 0.f;
        #pragma unroll
        for (int r = 0; r < 8; ++r) {
            p0[r] = __expf(l0[r] - nm); ps += p0[r];
            p1[r] = __expf(l1[r] - nm); ps += p1[r];
        }
        ps += __shfl_xor(ps, 16, 32);
        lsum = lsum * scale + ps;
        m = nm;

        // ---- rescale accumulator (acc row = r + 8*hi; stats live at lane r+8*hi) ----
        #pragma unroll
        for (int r = 0; r < 8; ++r) {
            float sc = __shfl(scale, r + 8 * hi, 32);
            #pragma unroll
            for (int j = 0; j < 16; ++j) acc[j][r] *= sc;
        }

        // ---- P already in A-operand layout: pack f32 -> bf16 ----
        BF16x16 ap;
        #pragma unroll
        for (int w = 0; w < 8; ++w) {
            float a = (w < 4) ? p0[2*w]     : p1[2*w - 8];
            float b = (w < 4) ? p0[2*w + 1] : p1[2*w - 7];
            ap.w[w] = (u32)f32_to_bf16(a) | ((u32)f32_to_bf16(b) << 16);
        }

        // ---- GEMM2: acc += P (16x32) x E_chunk (32x16 per dim-tile) ----
        #pragma unroll
        for (int j = 0; j < 16; ++j) {
            BF16x16 b2;
            const uint4* pb = (const uint4*)(trb + (16 * j + col) * CHUNK + 8 * hi);
            b2.q[0] = pb[0]; b2.q[1] = pb[2];
            acc[j] = __builtin_amdgcn_wmma_f32_16x16x32_bf16(false, ap.v, false, b2.v,
                                                             (short)0, acc[j], false, false);
        }
    }

    // ---- normalize, store z_soft, accumulate commitment loss ----
    float invl = 1.0f / lsum;
    float lossp = 0.f;
    #pragma unroll
    for (int r = 0; r < 8; ++r) {
        float il = __shfl(invl, r + 8 * hi, 32);
        int rl = wave * 16 + r + 8 * hi;
        size_t rg = (size_t)blockIdx.x * ROWS_PER_WG + rl;
        #pragma unroll
        for (int j = 0; j < 16; ++j) {
            int d = 16 * j + col;
            float v = acc[j][r] * il;
            out[rg * DIM + d] = v;
            float df = v - bf16_to_f32(zbf[rl * DIM + d]);
            lossp += df * df;
        }
    }
    #pragma unroll
    for (int off = 16; off >= 1; off >>= 1) lossp += __shfl_xor(lossp, off, 32);
    if (lane == 0)
        atomicAdd(loss, lossp * (1.0f / ((float)N_ROWS * (float)DIM)));
}

extern "C" void kernel_launch(void* const* d_in, const int* in_sizes, int n_in,
                              void* d_out, int out_size, void* d_ws, size_t ws_size,
                              hipStream_t stream) {
    const float* z  = (const float*)d_in[0];
    const float* cb = (const float*)d_in[1];
    float* out   = (float*)d_out;
    float* lossp = out + (size_t)N_ROWS * DIM;      // scalar tail of tuple output
    char*  ws    = (char*)d_ws;                     // ~1.03 MB scratch

    hipLaunchKernelGGL(svq_prep_kernel, dim3(4), dim3(256), 0, stream, cb, ws, lossp);
    hipLaunchKernelGGL(svq_main_kernel, dim3(N_ROWS / ROWS_PER_WG), dim3(256), LDS_BYTES,
                       stream, z, ws, out, lossp);
}
